// LinearBlock_38517266711179
// MI455X (gfx1250) — matrix operand order repaired and checked by reading, not compiled, not run on hardware
//
#include <hip/hip_runtime.h>
#include <stdint.h>

// Problem constants (match reference)
#define BATCH   8192
#define IN_CH   4096
#define OUT_CH  4096
#define QBLK    512     // IN_CH / 8

typedef __attribute__((ext_vector_type(16))) __bf16        v16bf;
typedef __attribute__((ext_vector_type(8)))  float         v8f;
typedef __attribute__((ext_vector_type(4)))  unsigned int  u32x4;
typedef __attribute__((ext_vector_type(2)))  unsigned int  u32x2;
typedef __attribute__((ext_vector_type(4)))  float         f32x4;

static __device__ __forceinline__ unsigned int f32_to_bf16_bits(float f) {
    union { float f; unsigned int u; } v; v.f = f;
    unsigned int u = v.u;
    // round-to-nearest-even
    u += 0x7FFFu + ((u >> 16) & 1u);
    return u >> 16;
}

// ---------------------------------------------------------------------------
// Pass 1: x fp32 -> bf16 (4 elements / thread). 8192*4096/4 threads, exact grid.
// ---------------------------------------------------------------------------
__global__ __launch_bounds__(256)
void cvt_x_bf16(const float* __restrict__ x, unsigned short* __restrict__ xb) {
    const int i = blockIdx.x * blockDim.x + threadIdx.x;   // < 8,388,608
    f32x4 v = reinterpret_cast<const f32x4*>(x)[i];
    u32x2 o;
    o.x = f32_to_bf16_bits(v.x) | (f32_to_bf16_bits(v.y) << 16);
    o.y = f32_to_bf16_bits(v.z) | (f32_to_bf16_bits(v.w) << 16);
    reinterpret_cast<u32x2*>(xb)[i] = o;
}

// ---------------------------------------------------------------------------
// Pass 2: expand circulant blocks into dense bf16 W, row-major [OUT_CH][IN_CH].
// W[o][i] = w[o>>3][i>>3][((o&7) - (i&7)) & 7]
// One thread per (o, q): reads 8 floats, writes 8 bf16 (16B).
// ---------------------------------------------------------------------------
__global__ __launch_bounds__(256)
void expand_w_bf16(const float* __restrict__ w, unsigned short* __restrict__ wexp) {
    const int t  = blockIdx.x * blockDim.x + threadIdx.x;  // < 4096*512
    const int o  = t >> 9;           // output channel
    const int q  = t & (QBLK - 1);   // input block
    const int p  = o >> 3;
    const int io = o & 7;
    const float* wc = w + (((size_t)p * QBLK + q) << 3);

    unsigned int r[8];
#pragma unroll
    for (int j = 0; j < 8; ++j)
        r[j] = f32_to_bf16_bits(wc[(io - j) & 7]);

    u32x4 out;
    out.x = r[0] | (r[1] << 16);
    out.y = r[2] | (r[3] << 16);
    out.z = r[4] | (r[5] << 16);
    out.w = r[6] | (r[7] << 16);
    *reinterpret_cast<u32x4*>(wexp + (size_t)o * IN_CH + (q << 3)) = out;
}

// ---------------------------------------------------------------------------
// Pass 3: y = clip(xb · Wexp^T, -1, 1) via v_wmma_f32_16x16x32_bf16.
// Workgroup: 256 threads = 8 waves as 2(M) x 4(N); wave tile 64x64.
// WG tile: 128(M) x 256(N). Grid: (8192/128, 4096/256) = (64, 16).
//
// Fragment layouts per CDNA5 ISA 7.12.2 (16-bit, wave32):
//   A (16x32, MxK): lane L holds row (L&15); elems 0..7  = K (L>>4)*8 + 0..7,
//                   elems 8..15 = K 16 + (L>>4)*8 + 0..7  -> two 16B loads.
//   B (32x16, KxN): lane L holds col (L&15); elems 0..7  = K (L>>4)*8 + 0..7,
//                   elems 8..15 = K 16 + (L>>4)*8 + 0..7  -> two 16B loads
//                   since Wexp is row-major by output channel (= column of B).
//   C (16x16 f32):  lane L holds col (L&15); vgpr r = row (L>>4)*8 + r.
// ---------------------------------------------------------------------------
__global__ __launch_bounds__(256)
void bcirc_gemm(const unsigned short* __restrict__ xb,
                const unsigned short* __restrict__ wb,
                float* __restrict__ y) {
    const int lane = threadIdx.x & 31;
    const int wid  = threadIdx.x >> 5;
    const int wm   = wid & 1;          // 2 waves along M
    const int wn   = wid >> 1;         // 4 waves along N
    const int m0   = blockIdx.x * 128 + wm * 64;
    const int n0   = blockIdx.y * 256 + wn * 64;
    const int l15  = lane & 15;
    const int sel  = lane >> 4;

    v8f c[4][4];
#pragma unroll
    for (int i = 0; i < 4; ++i)
#pragma unroll
        for (int j = 0; j < 4; ++j)
#pragma unroll
            for (int e = 0; e < 8; ++e) c[i][j][e] = 0.0f;

    const unsigned short* aBase = xb + (size_t)(m0 + l15) * IN_CH + sel * 8;
    const unsigned short* bBase = wb + (size_t)(n0 + l15) * IN_CH + sel * 8;

    for (int k0 = 0; k0 < IN_CH; k0 += 32) {
        union Frag { v16bf v; u32x4 q[2]; } a[4], b[4];

#pragma unroll
        for (int mt = 0; mt < 4; ++mt) {
            const unsigned short* p = aBase + (size_t)mt * 16 * IN_CH + k0;
            a[mt].q[0] = *reinterpret_cast<const u32x4*>(p);
            a[mt].q[1] = *reinterpret_cast<const u32x4*>(p + 16);
        }
#pragma unroll
        for (int nt = 0; nt < 4; ++nt) {
            const unsigned short* p = bBase + (size_t)nt * 16 * IN_CH + k0;
            b[nt].q[0] = *reinterpret_cast<const u32x4*>(p);
            b[nt].q[1] = *reinterpret_cast<const u32x4*>(p + 16);
        }

#pragma unroll
        for (int mt = 0; mt < 4; ++mt)
#pragma unroll
            for (int nt = 0; nt < 4; ++nt)
                c[mt][nt] = __builtin_amdgcn_wmma_f32_16x16x32_bf16(
                    /*neg_a=*/false, a[mt].v,
                    /*neg_b=*/false, b[nt].v,
                    /*c_mod=*/(short)0, c[mt][nt],
                    /*reuse_a=*/false, /*reuse_b=*/false);
    }

    // Epilogue: hardtanh(-1,1) via v_med3_f32 (1 VALU op), then fp32 store
    // (16 consecutive lanes -> 64B contiguous runs per half-wave).
#pragma unroll
    for (int mt = 0; mt < 4; ++mt) {
#pragma unroll
        for (int nt = 0; nt < 4; ++nt) {
            const int n = n0 + nt * 16 + l15;
            float* yp = y + (size_t)(m0 + mt * 16 + sel * 8) * OUT_CH + n;
#pragma unroll
            for (int r = 0; r < 8; ++r) {
                yp[(size_t)r * OUT_CH] =
                    __builtin_amdgcn_fmed3f(c[mt][nt][r], -1.0f, 1.0f);
            }
        }
    }
}

// ---------------------------------------------------------------------------
extern "C" void kernel_launch(void* const* d_in, const int* in_sizes, int n_in,
                              void* d_out, int out_size, void* d_ws, size_t ws_size,
                              hipStream_t stream) {
    (void)in_sizes; (void)n_in; (void)out_size; (void)ws_size;

    const float* x = (const float*)d_in[0];   // [8192, 4096] fp32
    const float* w = (const float*)d_in[1];   // [512, 512, 8] fp32
    float*       y = (float*)d_out;           // [8192, 4096] fp32

    // Workspace: xb (64 MB bf16) | wexp (32 MB bf16)  -> needs 96 MB of d_ws
    unsigned short* xb   = (unsigned short*)d_ws;
    unsigned short* wexp = xb + (size_t)BATCH * IN_CH;

    // Pass 1: x -> bf16
    cvt_x_bf16<<<(BATCH * (size_t)IN_CH) / 4 / 256, 256, 0, stream>>>(x, xb);
    // Pass 2: circulant expansion -> dense bf16 W [4096][4096]
    expand_w_bf16<<<((size_t)OUT_CH * QBLK) / 256, 256, 0, stream>>>(w, wexp);
    // Pass 3: WMMA GEMM + hardtanh
    dim3 grid(BATCH / 128, OUT_CH / 256);
    bcirc_gemm<<<grid, 256, 0, stream>>>(xb, wexp, y);
}
